// CSMv2Agent_68135361184310
// MI455X (gfx1250) — compile-verified, matrix operand order checked by reading, not run on hardware
//
#include <hip/hip_runtime.h>
#include <hip/hip_bf16.h>
#include <math.h>

typedef __attribute__((ext_vector_type(16))) _Float16 v16h;
typedef __attribute__((ext_vector_type(8)))  float    v8f;

#define BOBS 56
#define BH   64
#define BNA  6
#define BNSC 16
#define BSD  32
#define BMD  16
#define BGD  8
#define BNSLOT 4

// ---------------- f16 weight workspace layout (offsets in halves) -------------
// All matrices stored FRAGMENT-MAJOR: for each 32(K)x16(N) tile, lane l's 16
// halves are contiguous at [((kt*NTN + nt)*32 + l)*16 + i], matching the CDNA5
// WMMA B-operand VGPR layout, so a B fragment is one aligned 32-byte load.
enum {
  OFF_ENC = 0,                    // 64x64   (enc_W 56x64 zero-padded)
  OFF_WM  = OFF_ENC + 64*64,      // 96x64   (wm_W 70x64)
  OFF_MG  = OFF_WM  + 96*64,      // 96x16   (mg_W 67x8)
  OFF_RT  = OFF_MG  + 96*16,      // 96x16   (rt_W 72x16)
  OFF_EX  = OFF_RT  + 96*16,      // 16 x (64x64)
  OFF_MB  = OFF_EX  + 16*64*64,   // 64x32
  OFF_MC  = OFF_MB  + 64*32,      // 32x64
  OFF_MCW = OFF_MC  + 32*64,      // 64x16   rows 0..34 mc_in_W, 35..50 mc_rec_W
  OFF_WQ  = OFF_MCW + 64*16,      // 64x64
  OFF_WV  = OFF_WQ  + 64*64,      // 64x64
  OFF_CG  = OFF_WV  + 64*64,      // 64x16   (cg_W 64x4)
  OFF_C0  = OFF_CG  + 64*16,      // 64x16
  OFF_C1  = OFF_C0  + 64*16,      // 32x32
  OFF_C2  = OFF_C1  + 32*32,      // 96x64
  OFF_PIV = OFF_C2  + 96*64,      // 64x16   cols 0..5 pi_W, col 6 v_W
  WS_HALVES = OFF_PIV + 64*16
};

// ---------------- fast transcendentals (branch-free, TRANS-op based) ----------
// v_exp_f32 / v_log_f32 are TRANS32 ops that co-execute with the WMMA/XDL pipe,
// and v_rcp_f32 avoids the precise-division v_div_scale expansion. The data has
// already passed through f16 WMMA, so libm-exact tails are wasted issue slots.
__device__ inline float frcp(float x)  { return __builtin_amdgcn_rcpf(x); }
__device__ inline float fexp(float x)  { return __expf(x); }
__device__ inline float flog(float x)  { return __logf(x); }
__device__ inline float ftanh(float x) {
  float e = __expf(-2.f * fabsf(x));
  float t = (1.f - e) * frcp(1.f + e);
  return copysignf(t, x);
}
__device__ inline float fsigmoid(float x) { return frcp(1.f + __expf(-x)); }
__device__ inline float fsoftplus(float x) {
  // stable: max(x,0) + log1p(exp(-|x|)), branch-free
  return fmaxf(x, 0.f) + __logf(1.f + __expf(-fabsf(x)));
}

// ---------------- WMMA helpers (CDNA5 16x16x32 f16 -> f32) --------------------
__device__ inline v8f v8zero() {
  v8f z;
#pragma unroll
  for (int i = 0; i < 8; ++i) z[i] = 0.f;
  return z;
}

__device__ inline v8f wmma_f16(v16h a, v16h b, v8f c) {
  // D = A(16x32 f16) * B(32x16 f16) + C(16x16 f32)
  return __builtin_amdgcn_wmma_f32_16x16x32_f16(false, a, false, b, (short)0, c,
                                                false, false);
}

// A fragment: 16 rows x 32 K window from f16 row-major LDS buffer.
// Lane l (l<16): row l, K = k0+[0..7] and k0+[16..23]; lanes 16..31: +8 on K.
// (Each 8-half run is contiguous -> backend emits ds_load_b128.)
__device__ inline v16h load_a_frag(const _Float16* buf, int stride, int k0) {
  const int lane = threadIdx.x & 31;
  const _Float16* p = buf + (lane & 15) * stride + k0 + ((lane & 16) ? 8 : 0);
  v16h a;
#pragma unroll
  for (int g = 0; g < 2; ++g)
#pragma unroll
    for (int j = 0; j < 8; ++j)
      a[g * 8 + j] = p[g * 16 + j];
  return a;
}

// B fragment from fragment-major weights: one aligned 32-byte vector load.
__device__ inline v16h load_bf(const _Float16* Wf, int ntn, int kt, int nt) {
  const int lane = threadIdx.x & 31;
  return *(const v16h*)(Wf + ((((long)kt * ntn + nt) * 32 + lane) << 4));
}

// ---------------- weight prep kernel (f32 -> fragment-major f16, once) --------
template <typename F>
__device__ inline void frag_pack(_Float16* dst, int Kp, int Np, int tid,
                                 int nthr, F getv) {
  const int ntn = Np >> 4;
  const int total = Kp * Np;
  for (int idx = tid; idx < total; idx += nthr) {
    const int i = idx & 15;          // half index within lane
    const int l = (idx >> 4) & 31;   // lane
    const int t = idx >> 9;          // tile = kt*ntn + nt
    const int kt = t / ntn, nt = t - kt * ntn;
    const int n = nt * 16 + (l & 15);
    const int k = kt * 32 + ((l & 16) ? 8 : 0) + ((i < 8) ? i : (i + 8));
    dst[idx] = (_Float16)getv(k, n);
  }
}

struct PrepParams {
  const float *enc_W, *wm_W, *mg_W, *rt_W, *ex_W, *mB_W, *mC_W;
  const float *mc_in_W, *mc_rec_W, *Wq, *Wv, *cg_W, *c0_W, *c1_W, *c2_W;
  const float *pi_W, *v_W;
  _Float16* ws;
};

__global__ void csm_prep_weights(PrepParams p) {
  const int tid  = blockIdx.x * blockDim.x + threadIdx.x;
  const int nthr = gridDim.x * blockDim.x;
  _Float16* ws = p.ws;

  frag_pack(ws + OFF_ENC, 64, 64, tid, nthr, [=](int k, int n) {
    return (k < BOBS) ? p.enc_W[k * BH + n] : 0.f; });
  frag_pack(ws + OFF_WM, 96, 64, tid, nthr, [=](int k, int n) {
    return (k < BH + BNA) ? p.wm_W[k * BH + n] : 0.f; });
  frag_pack(ws + OFF_MG, 96, 16, tid, nthr, [=](int k, int n) {
    return (k < BH + 3 && n < BGD) ? p.mg_W[k * BGD + n] : 0.f; });
  frag_pack(ws + OFF_RT, 96, 16, tid, nthr, [=](int k, int n) {
    return (k < BH + BGD) ? p.rt_W[k * BNSC + n] : 0.f; });
  for (int s = 0; s < BNSC; ++s) {
    const float* src = p.ex_W + (long)s * BH * BH;
    frag_pack(ws + OFF_EX + s * 64 * 64, 64, 64, tid, nthr, [=](int k, int n) {
      return src[k * BH + n]; });
  }
  frag_pack(ws + OFF_MB, 64, 32, tid, nthr, [=](int k, int n) {
    return p.mB_W[k * BSD + n]; });
  frag_pack(ws + OFF_MC, 32, 64, tid, nthr, [=](int k, int n) {
    return p.mC_W[k * BH + n]; });
  // K-stacked metacognition weight: [S_obj(32), perr, conf, entr, S_meta(16)]->16
  frag_pack(ws + OFF_MCW, 64, 16, tid, nthr, [=](int k, int n) {
    if (k < BSD + 3)       return p.mc_in_W[k * BMD + n];
    if (k < BSD + 3 + BMD) return p.mc_rec_W[(k - (BSD + 3)) * BMD + n];
    return 0.f; });
  frag_pack(ws + OFF_WQ, 64, 64, tid, nthr, [=](int k, int n) {
    return p.Wq[k * BH + n]; });
  frag_pack(ws + OFF_WV, 64, 64, tid, nthr, [=](int k, int n) {
    return p.Wv[k * BH + n]; });
  frag_pack(ws + OFF_CG, 64, 16, tid, nthr, [=](int k, int n) {
    return (n < 4) ? p.cg_W[k * 4 + n] : 0.f; });
  frag_pack(ws + OFF_C0, 64, 16, tid, nthr, [=](int k, int n) {
    return p.c0_W[k * 16 + n]; });
  frag_pack(ws + OFF_C1, 32, 32, tid, nthr, [=](int k, int n) {
    return p.c1_W[k * BSD + n]; });
  frag_pack(ws + OFF_C2, 96, 64, tid, nthr, [=](int k, int n) {
    return p.c2_W[k * BH + n]; });
  // packed policy/value head: cols 0..5 = pi_W, col 6 = v_W
  frag_pack(ws + OFF_PIV, 64, 16, tid, nthr, [=](int k, int n) {
    if (n < BNA)  return p.pi_W[k * BNA + n];
    if (n == BNA) return p.v_W[k];
    return 0.f; });
}

// ---------------- fused forward kernel ---------------------------------------
struct FusedParams {
  const float *obs, *prev_h, *pact, *S_obj, *S_meta, *slots, *extr;
  const float *enc_b, *wm_b, *mg_b, *rt_b, *ex_b, *Avec, *mB_b, *mC_b, *mc_b;
  const float *dA_W, *dA_b, *ip_W, *ip_b, *cg_b, *c0_b, *c1_b, *c2_b, *pi_b, *v_b;
  const _Float16* ws;
  float* out;
  int B;
};

struct WaveLds {
  _Float16 stage[16 * 96];   // A-matrix staging (K padded to 96)
  _Float16 h16[16 * 64];     // h (encoder output)
  _Float16 hr16[16 * 64];    // h_routed, later reused for c2
  _Float16 qf[16 * 64];      // q = hr @ Wq
  _Float16 vf[16 * 64];      // v = hr @ Wv
  float sw[16 * 16];         // logits/scene_weights, smeta, piv (reused)
  float perr[16], conf[16], entr[16], dAv[16], capg[16];
};

__global__ __launch_bounds__(128, 1) void csm_fused_agent(FusedParams P) {
  __shared__ WaveLds lds[4];
  const int lane = threadIdx.x & 31;
  const int wave = threadIdx.x >> 5;
  WaveLds& L = lds[wave];
  const int rowBase = blockIdx.x * 64 + wave * 16;
  const int nl   = lane & 15;              // N-column within 16-tile
  const int moff = (lane & 16) ? 8 : 0;    // M offset for C/D rows
  const long B = (long)P.B;

  float* out_action = P.out;
  float* out_value  = P.out + 6   * B;
  float* out_goal   = P.out + 7   * B;
  float* out_sw     = P.out + 15  * B;
  float* out_interp = P.out + 31  * B;
  float* out_var    = P.out + 34  * B;
  float* out_c0     = P.out + 38  * B;
  float* out_c1     = P.out + 54  * B;
  float* out_c2     = P.out + 86  * B;
  float* out_h      = P.out + 150 * B;
  float* out_y      = P.out + 214 * B;
  float* out_sobj   = P.out + 278 * B;
  float* out_smeta  = P.out + 310 * B;
  float* out_slots  = P.out + 326 * B;

  // ============ Phase A: h = tanh(obs @ enc_W + b) ============
  for (int i = lane; i < 16 * 96; i += 32) L.stage[i] = (_Float16)0.f;
  for (int i = lane; i < 16 * BOBS; i += 32) {
    int r = i / BOBS, c = i % BOBS;
    L.stage[r * 96 + c] = (_Float16)P.obs[(long)(rowBase + r) * BOBS + c];
  }
  __syncthreads();
  {
    v16h a0 = load_a_frag(L.stage, 96, 0);
    v16h a1 = load_a_frag(L.stage, 96, 32);
#pragma unroll
    for (int nt = 0; nt < 4; ++nt) {
      v8f d = v8zero();
      d = wmma_f16(a0, load_bf(P.ws + OFF_ENC, 4, 0, nt), d);
      d = wmma_f16(a1, load_bf(P.ws + OFF_ENC, 4, 1, nt), d);
      const int n = nt * 16 + nl;
      const float bias = P.enc_b[n];
#pragma unroll
      for (int r = 0; r < 8; ++r) {
        float v = ftanh(d[r] + bias);
        out_h[(long)(rowBase + r + moff) * 64 + n] = v;
        L.h16[(r + moff) * 64 + n] = (_Float16)v;
      }
    }
  }
  __syncthreads();

  // ============ Phase A2: c0 and var_probs (both from h) ============
  {
    v16h ah0 = load_a_frag(L.h16, 64, 0);
    v16h ah1 = load_a_frag(L.h16, 64, 32);
    {
      v8f d = v8zero();
      d = wmma_f16(ah0, load_bf(P.ws + OFF_C0, 1, 0, 0), d);
      d = wmma_f16(ah1, load_bf(P.ws + OFF_C0, 1, 1, 0), d);
      const float bias = P.c0_b[nl];
#pragma unroll
      for (int r = 0; r < 8; ++r)
        out_c0[(long)(rowBase + r + moff) * 16 + nl] = d[r] + bias;
    }
    {
      v8f d = v8zero();
      d = wmma_f16(ah0, load_bf(P.ws + OFF_CG, 1, 0, 0), d);
      d = wmma_f16(ah1, load_bf(P.ws + OFF_CG, 1, 1, 0), d);
      if (nl < 4) {
        const float bias = P.cg_b[nl];
#pragma unroll
        for (int r = 0; r < 8; ++r)
          out_var[(long)(rowBase + r + moff) * 4 + nl] = fsigmoid(d[r] + bias);
      }
    }
  }

  // ============ Phase B: pred_err from world model ============
  for (int i = lane; i < 16 * 96; i += 32) L.stage[i] = (_Float16)0.f;
  for (int i = lane; i < 16 * 64; i += 32) {
    int r = i >> 6, c = i & 63;
    L.stage[r * 96 + c] = (_Float16)P.prev_h[(long)(rowBase + r) * 64 + c];
  }
  for (int i = lane; i < 16 * BNA; i += 32) {
    int r = i / BNA, c = i % BNA;
    L.stage[r * 96 + 64 + c] = (_Float16)P.pact[(long)(rowBase + r) * BNA + c];
  }
  if (lane < 16) L.perr[lane] = 0.f;
  __syncthreads();
  {
    v16h a0 = load_a_frag(L.stage, 96, 0);
    v16h a1 = load_a_frag(L.stage, 96, 32);
    v16h a2 = load_a_frag(L.stage, 96, 64);
    float s[8] = {0.f, 0.f, 0.f, 0.f, 0.f, 0.f, 0.f, 0.f};
#pragma unroll
    for (int nt = 0; nt < 4; ++nt) {
      v8f d = v8zero();
      d = wmma_f16(a0, load_bf(P.ws + OFF_WM, 4, 0, nt), d);
      d = wmma_f16(a1, load_bf(P.ws + OFF_WM, 4, 1, nt), d);
      d = wmma_f16(a2, load_bf(P.ws + OFF_WM, 4, 2, nt), d);
      const int n = nt * 16 + nl;
      const float bias = P.wm_b[n];
#pragma unroll
      for (int r = 0; r < 8; ++r) {
        float diff = d[r] + bias - (float)L.h16[(r + moff) * 64 + n];
        s[r] += diff * diff;
      }
    }
#pragma unroll
    for (int r = 0; r < 8; ++r) atomicAdd(&L.perr[r + moff], s[r]);
  }
  __syncthreads();
  if (lane < 16) {
    const int m = lane;
    float pe = L.perr[m] * (1.f / 64.f);
    L.perr[m] = pe;
    L.conf[m] = 1.f - fminf(pe, 1.f);
    float er = P.extr[rowBase + m];
    L.capg[m] = (er < 0.f) ? -er : 0.1f;
  }
  __syncthreads();

  // ============ Phase C: goal = tanh([h, 0, capgap, extr] @ mg_W + b) ============
  for (int i = lane; i < 16 * 96; i += 32) L.stage[i] = (_Float16)0.f;
  for (int i = lane; i < 16 * 64; i += 32) {
    int r = i >> 6, c = i & 63;
    L.stage[r * 96 + c] = L.h16[i];
  }
  if (lane < 16) {
    L.stage[lane * 96 + 65] = (_Float16)L.capg[lane];
    L.stage[lane * 96 + 66] = (_Float16)P.extr[rowBase + lane];
  }
  __syncthreads();
  {
    v16h a0 = load_a_frag(L.stage, 96, 0);
    v16h a1 = load_a_frag(L.stage, 96, 32);
    v16h a2 = load_a_frag(L.stage, 96, 64);
    v8f d = v8zero();
    d = wmma_f16(a0, load_bf(P.ws + OFF_MG, 1, 0, 0), d);
    d = wmma_f16(a1, load_bf(P.ws + OFF_MG, 1, 1, 0), d);
    d = wmma_f16(a2, load_bf(P.ws + OFF_MG, 1, 2, 0), d);
    if (nl < BGD) {
      const float bias = P.mg_b[nl];
#pragma unroll
      for (int r = 0; r < 8; ++r) {
        float g = ftanh(d[r] + bias);
        out_goal[(long)(rowBase + r + moff) * BGD + nl] = g;
        // append goal to stage cols 64..71 -> router input [h, goal]
        L.stage[(r + moff) * 96 + 64 + nl] = (_Float16)g;
      }
    }
  }
  __syncthreads();

  // ============ Phase D: router logits, top-2 softmax scatter ============
  {
    v16h a0 = load_a_frag(L.stage, 96, 0);
    v16h a1 = load_a_frag(L.stage, 96, 32);
    v16h a2 = load_a_frag(L.stage, 96, 64);
    v8f d = v8zero();
    d = wmma_f16(a0, load_bf(P.ws + OFF_RT, 1, 0, 0), d);
    d = wmma_f16(a1, load_bf(P.ws + OFF_RT, 1, 1, 0), d);
    d = wmma_f16(a2, load_bf(P.ws + OFF_RT, 1, 2, 0), d);
    const float bias = P.rt_b[nl];
#pragma unroll
    for (int r = 0; r < 8; ++r) L.sw[(r + moff) * 16 + nl] = d[r] + bias;
  }
  __syncthreads();
  if (lane < 16) {
    const int m = lane;
    float l[16];
#pragma unroll
    for (int j = 0; j < 16; ++j) l[j] = L.sw[m * 16 + j];
    int i0 = 0;
#pragma unroll
    for (int j = 1; j < 16; ++j) if (l[j] > l[i0]) i0 = j;
    int i1 = (i0 == 0) ? 1 : 0;
#pragma unroll
    for (int j = 0; j < 16; ++j) if (j != i0 && l[j] > l[i1]) i1 = j;
    float e1 = fexp(l[i1] - l[i0]);
    float inv = frcp(1.f + e1);
    float w0 = inv, w1 = e1 * inv;
#pragma unroll
    for (int j = 0; j < 16; ++j) {
      float w = (j == i0) ? w0 : ((j == i1) ? w1 : 0.f);
      L.sw[m * 16 + j] = w;
      out_sw[(long)(rowBase + m) * 16 + j] = w;
    }
    L.entr[m] = -(w0 * flog(w0 + 1e-8f) + w1 * flog(w1 + 1e-8f));
  }
  __syncthreads();

  // ============ Phase E: 16 experts, weighted-relu accumulation ============
  {
    v16h ah0 = load_a_frag(L.h16, 64, 0);
    v16h ah1 = load_a_frag(L.h16, 64, 32);
    v8f hr[4];
#pragma unroll
    for (int nt = 0; nt < 4; ++nt) hr[nt] = v8zero();
    for (int s = 0; s < BNSC; ++s) {
      const _Float16* We = P.ws + OFF_EX + s * 64 * 64;
      if (s + 1 < BNSC)
        __builtin_prefetch((const void*)(We + 64 * 64), 0, 0);  // global_prefetch_b8
      float wr[8];
#pragma unroll
      for (int r = 0; r < 8; ++r) wr[r] = L.sw[(r + moff) * 16 + s];
#pragma unroll
      for (int nt = 0; nt < 4; ++nt) {
        v8f t = v8zero();
        t = wmma_f16(ah0, load_bf(We, 4, 0, nt), t);
        t = wmma_f16(ah1, load_bf(We, 4, 1, nt), t);
        const float bb = P.ex_b[s * 64 + nt * 16 + nl];
#pragma unroll
        for (int r = 0; r < 8; ++r) {
          float e = fmaxf(t[r] + bb, 0.f);
          hr[nt][r] += wr[r] * e;
        }
      }
    }
#pragma unroll
    for (int nt = 0; nt < 4; ++nt) {
      const int n = nt * 16 + nl;
#pragma unroll
      for (int r = 0; r < 8; ++r)
        L.hr16[(r + moff) * 64 + n] = (_Float16)hr[nt][r];
    }
  }
  __syncthreads();

  // ============ Phase F: metacognition S_meta_new, delta_A, interp ============
  for (int i = lane; i < 16 * 96; i += 32) L.stage[i] = (_Float16)0.f;
  for (int i = lane; i < 16 * BSD; i += 32) {
    int r = i >> 5, c = i & 31;
    L.stage[r * 96 + c] = (_Float16)P.S_obj[(long)(rowBase + r) * BSD + c];
  }
  for (int i = lane; i < 16 * BMD; i += 32) {
    int r = i >> 4, c = i & 15;
    L.stage[r * 96 + 35 + c] = (_Float16)P.S_meta[(long)(rowBase + r) * BMD + c];
  }
  if (lane < 16) {
    L.stage[lane * 96 + 32] = (_Float16)L.perr[lane];
    L.stage[lane * 96 + 33] = (_Float16)L.conf[lane];
    L.stage[lane * 96 + 34] = (_Float16)L.entr[lane];
  }
  __syncthreads();
  {
    v16h a0 = load_a_frag(L.stage, 96, 0);
    v16h a1 = load_a_frag(L.stage, 96, 32);
    v8f d = v8zero();
    d = wmma_f16(a0, load_bf(P.ws + OFF_MCW, 1, 0, 0), d);
    d = wmma_f16(a1, load_bf(P.ws + OFF_MCW, 1, 1, 0), d);
    const float bias = P.mc_b[nl];
#pragma unroll
    for (int r = 0; r < 8; ++r) {
      float v = ftanh(d[r] + bias);
      out_smeta[(long)(rowBase + r + moff) * 16 + nl] = v;
      L.sw[(r + moff) * 16 + nl] = v;
    }
  }
  __syncthreads();
  if (lane < 16) {
    const int m = lane;
    float acc = P.dA_b[0];
    float o0 = P.ip_b[0], o1 = P.ip_b[1], o2 = P.ip_b[2];
#pragma unroll
    for (int j = 0; j < 16; ++j) {
      float sv = L.sw[m * 16 + j];
      acc += sv * P.dA_W[j];
      o0 += sv * P.ip_W[j * 3 + 0];
      o1 += sv * P.ip_W[j * 3 + 1];
      o2 += sv * P.ip_W[j * 3 + 2];
    }
    L.dAv[m] = fsoftplus(acc);
    float mx = fmaxf(o0, fmaxf(o1, o2));
    float e0 = fexp(o0 - mx), e1 = fexp(o1 - mx), e2 = fexp(o2 - mx);
    float inv = frcp(e0 + e1 + e2);
    out_interp[(long)(rowBase + m) * 3 + 0] = e0 * inv;
    out_interp[(long)(rowBase + m) * 3 + 1] = e1 * inv;
    out_interp[(long)(rowBase + m) * 3 + 2] = e2 * inv;
  }
  __syncthreads();

  // ============ Phase G: SSM step S_obj_new ============
  {
    v16h ar0 = load_a_frag(L.hr16, 64, 0);
    v16h ar1 = load_a_frag(L.hr16, 64, 32);
#pragma unroll
    for (int nt = 0; nt < 2; ++nt) {
      v8f d = v8zero();
      d = wmma_f16(ar0, load_bf(P.ws + OFF_MB, 2, 0, nt), d);
      d = wmma_f16(ar1, load_bf(P.ws + OFF_MB, 2, 1, nt), d);
      const int n = nt * 16 + nl;
      const float bias = P.mB_b[n];
      const float sp = fsoftplus(P.Avec[n]);
#pragma unroll
      for (int r = 0; r < 8; ++r) {
        const int m = r + moff;
        float dec = fexp(-sp * L.dAv[m]);
        float so = dec * P.S_obj[(long)(rowBase + m) * BSD + n] + d[r] + bias;
        out_sobj[(long)(rowBase + m) * BSD + n] = so;
        L.stage[m * 96 + n] = (_Float16)so;   // c2/c1/y input, cols 0..31
      }
    }
  }
  __syncthreads();

  // ============ Phase H: y = S_obj_new@mC+b, c1 = S_obj_new@c1_W+b ============
  {
    v16h as = load_a_frag(L.stage, 96, 0);   // K = 32
#pragma unroll
    for (int nt = 0; nt < 4; ++nt) {
      v8f d = v8zero();
      d = wmma_f16(as, load_bf(P.ws + OFF_MC, 4, 0, nt), d);
      const int n = nt * 16 + nl;
      const float bias = P.mC_b[n];
#pragma unroll
      for (int r = 0; r < 8; ++r)
        out_y[(long)(rowBase + r + moff) * 64 + n] = d[r] + bias;
    }
#pragma unroll
    for (int nt = 0; nt < 2; ++nt) {
      v8f d = v8zero();
      d = wmma_f16(as, load_bf(P.ws + OFF_C1, 2, 0, nt), d);
      const int n = nt * 16 + nl;
      const float bias = P.c1_b[n];
#pragma unroll
      for (int r = 0; r < 8; ++r)
        out_c1[(long)(rowBase + r + moff) * BSD + n] = d[r] + bias;
    }
  }

  // ============ Phase I: q/v projections + slot attention ============
  {
    v16h ar0 = load_a_frag(L.hr16, 64, 0);
    v16h ar1 = load_a_frag(L.hr16, 64, 32);
#pragma unroll
    for (int nt = 0; nt < 4; ++nt) {
      v8f q = v8zero();
      q = wmma_f16(ar0, load_bf(P.ws + OFF_WQ, 4, 0, nt), q);
      q = wmma_f16(ar1, load_bf(P.ws + OFF_WQ, 4, 1, nt), q);
      v8f v = v8zero();
      v = wmma_f16(ar0, load_bf(P.ws + OFF_WV, 4, 0, nt), v);
      v = wmma_f16(ar1, load_bf(P.ws + OFF_WV, 4, 1, nt), v);
      const int n = nt * 16 + nl;
#pragma unroll
      for (int r = 0; r < 8; ++r) {
        L.qf[(r + moff) * 64 + n] = (_Float16)q[r];
        L.vf[(r + moff) * 64 + n] = (_Float16)v[r];
      }
    }
  }
  __syncthreads();
  if (lane < 16) {
    const int m = lane;
    const float* srow = P.slots + (long)(rowBase + m) * BNSLOT * 64;
    float* drow = out_slots + (long)(rowBase + m) * BNSLOT * 64;
    float dots[BNSLOT];
#pragma unroll
    for (int n = 0; n < BNSLOT; ++n) {
      float a = 0.f;
      for (int k = 0; k < 64; ++k) a += srow[n * 64 + k] * (float)L.qf[m * 64 + k];
      dots[n] = a * 0.125f;  // 1/sqrt(64)
    }
    float mx = dots[0];
#pragma unroll
    for (int n = 1; n < BNSLOT; ++n) mx = fmaxf(mx, dots[n]);
    float attn[BNSLOT], sum = 0.f;
#pragma unroll
    for (int n = 0; n < BNSLOT; ++n) { attn[n] = fexp(dots[n] - mx); sum += attn[n]; }
    float inv = frcp(sum);
#pragma unroll
    for (int n = 0; n < BNSLOT; ++n) attn[n] *= inv;
    for (int k = 0; k < 64; ++k) {
      float vv = (float)L.vf[m * 64 + k];
      float sr = 0.f;
#pragma unroll
      for (int n = 0; n < BNSLOT; ++n) {
        float sl = srow[n * 64 + k];
        sr += attn[n] * sl;
        drow[n * 64 + k] = sl + attn[n] * vv;
      }
      L.stage[m * 96 + 32 + k] = (_Float16)sr;   // slot_read -> c2 input cols 32..95
    }
  }
  __syncthreads();

  // ============ Phase J: c2 = [S_obj_new, slot_read] @ c2_W + b ============
  {
    v16h a0 = load_a_frag(L.stage, 96, 0);
    v16h a1 = load_a_frag(L.stage, 96, 32);
    v16h a2 = load_a_frag(L.stage, 96, 64);
#pragma unroll
    for (int nt = 0; nt < 4; ++nt) {
      v8f d = v8zero();
      d = wmma_f16(a0, load_bf(P.ws + OFF_C2, 4, 0, nt), d);
      d = wmma_f16(a1, load_bf(P.ws + OFF_C2, 4, 1, nt), d);
      d = wmma_f16(a2, load_bf(P.ws + OFF_C2, 4, 2, nt), d);
      const int n = nt * 16 + nl;
      const float bias = P.c2_b[n];
#pragma unroll
      for (int r = 0; r < 8; ++r) {
        float v = d[r] + bias;
        out_c2[(long)(rowBase + r + moff) * 64 + n] = v;
        L.hr16[(r + moff) * 64 + n] = (_Float16)v;   // reuse as c2 f16
      }
    }
  }
  __syncthreads();

  // ============ Phase K: packed policy/value head + softmax ============
  {
    v16h ac0 = load_a_frag(L.hr16, 64, 0);
    v16h ac1 = load_a_frag(L.hr16, 64, 32);
    v8f d = v8zero();
    d = wmma_f16(ac0, load_bf(P.ws + OFF_PIV, 1, 0, 0), d);
    d = wmma_f16(ac1, load_bf(P.ws + OFF_PIV, 1, 1, 0), d);
    const float bias = (nl < BNA) ? P.pi_b[nl] : ((nl == BNA) ? P.v_b[0] : 0.f);
#pragma unroll
    for (int r = 0; r < 8; ++r) L.sw[(r + moff) * 16 + nl] = d[r] + bias;
  }
  __syncthreads();
  if (lane < 16) {
    const int m = lane;
    float lg[BNA];
    float mx = -1e30f;
#pragma unroll
    for (int j = 0; j < BNA; ++j) { lg[j] = L.sw[m * 16 + j]; mx = fmaxf(mx, lg[j]); }
    float sum = 0.f;
#pragma unroll
    for (int j = 0; j < BNA; ++j) { lg[j] = fexp(lg[j] - mx); sum += lg[j]; }
    float inv = frcp(sum);
#pragma unroll
    for (int j = 0; j < BNA; ++j)
      out_action[(long)(rowBase + m) * BNA + j] = lg[j] * inv;
    out_value[rowBase + m] = L.sw[m * 16 + BNA];
  }
}

// ---------------- launcher ----------------------------------------------------
extern "C" void kernel_launch(void* const* d_in, const int* in_sizes, int n_in,
                              void* d_out, int out_size, void* d_ws, size_t ws_size,
                              hipStream_t stream) {
  const float* obs     = (const float*)d_in[0];
  const float* prev_h  = (const float*)d_in[1];
  const float* pact    = (const float*)d_in[2];
  const float* S_obj   = (const float*)d_in[3];
  const float* S_meta  = (const float*)d_in[4];
  const float* slots   = (const float*)d_in[5];
  const float* extr    = (const float*)d_in[6];
  const float* enc_W   = (const float*)d_in[7];
  const float* enc_b   = (const float*)d_in[8];
  const float* mg_W    = (const float*)d_in[9];
  const float* mg_b    = (const float*)d_in[10];
  const float* rt_W    = (const float*)d_in[11];
  const float* rt_b    = (const float*)d_in[12];
  const float* ex_W    = (const float*)d_in[13];
  const float* ex_b    = (const float*)d_in[14];
  const float* Avec    = (const float*)d_in[15];
  const float* mB_W    = (const float*)d_in[16];
  const float* mB_b    = (const float*)d_in[17];
  const float* mC_W    = (const float*)d_in[18];
  const float* mC_b    = (const float*)d_in[19];
  const float* mc_in_W = (const float*)d_in[20];
  const float* mc_in_b = (const float*)d_in[21];
  const float* mc_rec_W= (const float*)d_in[22];
  const float* dA_W    = (const float*)d_in[23];
  const float* dA_b    = (const float*)d_in[24];
  const float* ip_W    = (const float*)d_in[25];
  const float* ip_b    = (const float*)d_in[26];
  const float* Wq      = (const float*)d_in[27];
  const float* Wv      = (const float*)d_in[28];
  const float* cg_W    = (const float*)d_in[29];
  const float* cg_b    = (const float*)d_in[30];
  const float* c0_W    = (const float*)d_in[31];
  const float* c0_b    = (const float*)d_in[32];
  const float* c1_W    = (const float*)d_in[33];
  const float* c1_b    = (const float*)d_in[34];
  const float* c2_W    = (const float*)d_in[35];
  const float* c2_b    = (const float*)d_in[36];
  const float* wm_W    = (const float*)d_in[37];
  const float* wm_b    = (const float*)d_in[38];
  const float* pi_W    = (const float*)d_in[39];
  const float* pi_b    = (const float*)d_in[40];
  const float* v_W     = (const float*)d_in[41];
  const float* v_b     = (const float*)d_in[42];

  _Float16* ws = (_Float16*)d_ws;
  const int Bn = in_sizes[0] / BOBS;

  PrepParams pp;
  pp.enc_W = enc_W; pp.wm_W = wm_W; pp.mg_W = mg_W; pp.rt_W = rt_W;
  pp.ex_W = ex_W; pp.mB_W = mB_W; pp.mC_W = mC_W; pp.mc_in_W = mc_in_W;
  pp.mc_rec_W = mc_rec_W; pp.Wq = Wq; pp.Wv = Wv; pp.cg_W = cg_W;
  pp.c0_W = c0_W; pp.c1_W = c1_W; pp.c2_W = c2_W; pp.pi_W = pi_W; pp.v_W = v_W;
  pp.ws = ws;
  csm_prep_weights<<<64, 256, 0, stream>>>(pp);

  FusedParams fp;
  fp.obs = obs; fp.prev_h = prev_h; fp.pact = pact; fp.S_obj = S_obj;
  fp.S_meta = S_meta; fp.slots = slots; fp.extr = extr;
  fp.enc_b = enc_b; fp.wm_b = wm_b; fp.mg_b = mg_b; fp.rt_b = rt_b;
  fp.ex_b = ex_b; fp.Avec = Avec; fp.mB_b = mB_b; fp.mC_b = mC_b;
  fp.mc_b = mc_in_b; fp.dA_W = dA_W; fp.dA_b = dA_b; fp.ip_W = ip_W;
  fp.ip_b = ip_b; fp.cg_b = cg_b; fp.c0_b = c0_b; fp.c1_b = c1_b;
  fp.c2_b = c2_b; fp.pi_b = pi_b; fp.v_b = v_b;
  fp.ws = ws; fp.out = (float*)d_out; fp.B = Bn;
  csm_fused_agent<<<Bn / 64, 128, 0, stream>>>(fp);
}
